// CrossAttention_47596827574651
// MI455X (gfx1250) — matrix operand order repaired and checked by reading, not compiled, not run on hardware
//
#include <hip/hip_runtime.h>
#include <hip/hip_bf16.h>

typedef __attribute__((ext_vector_type(16))) _Float16 v16h;
typedef __attribute__((ext_vector_type(8)))  float    v8f;
typedef __attribute__((ext_vector_type(4)))  int      v4i;

#define NB    16
#define INCH  256
#define EMBD  256
#define CTXD  128
#define SEQ   2048
#define HW    2304   // 48*48

#define WMMA_F16(A, B, C) \
  __builtin_amdgcn_wmma_f32_16x16x32_f16(false, (A), false, (B), (short)0, (C), false, false)

// ---------------- async global->LDS staging (CDNA5 ASYNCcnt path) ----------------
#if defined(__gfx1250__) && __has_builtin(__builtin_amdgcn_global_load_async_to_lds_b128) && \
    __has_builtin(__builtin_amdgcn_s_wait_asynccnt)
#define HAVE_ASYNC_LDS 1
#else
#define HAVE_ASYNC_LDS 0
#endif

// Copy `bytes` (multiple of 16) from global to LDS cooperatively.
static __device__ __forceinline__ void stage_to_lds(const void* gsrc, void* ldst,
                                                    int tid, int nthr, int bytes) {
#if HAVE_ASYNC_LDS
  typedef char __attribute__((address_space(1))) gch;
  typedef char __attribute__((address_space(3))) lch;
  typedef v4i  __attribute__((address_space(1))) gvec;
  typedef v4i  __attribute__((address_space(3))) lvec;
  gch* g = (gch*)gsrc;
  lch* l = (lch*)ldst;
  for (int o = tid * 16; o < bytes; o += nthr * 16)
    __builtin_amdgcn_global_load_async_to_lds_b128((gvec*)(g + o), (lvec*)(l + o), 0, 0);
#else
  const uint4* g = (const uint4*)gsrc;
  uint4* l = (uint4*)ldst;
  for (int o = tid; o < bytes / 16; o += nthr) l[o] = g[o];
#endif
}
static __device__ __forceinline__ void stage_wait() {
#if HAVE_ASYNC_LDS
  __builtin_amdgcn_s_wait_asynccnt(0);
#endif
}

// ---------------- fragment helpers (CDNA5 wave32 WMMA layouts) ----------------
// A (16x32 f16): lane m = lane&15, half = lane>>4; VGPR j holds K pair:
//   K = (j<4 ? 0 : 16) + 8*half + 2*(j&3)
// B (32x16) loaded from B^T row-major ([N x K]) with identical indexing.
// C/D (16x16 f32): col n = lane&15, VGPR i holds row m = i + 8*half.

static __device__ __forceinline__ v16h frag_ld_f16(const _Float16* __restrict__ p,
                                                   int ld, int lane) {
  const int m = lane & 15, h = lane >> 4;
  const _Float16* r = p + (size_t)m * ld;
  v16h a;
#pragma unroll
  for (int j = 0; j < 8; ++j) {
    const int k = ((j < 4) ? 0 : 16) + 8 * h + 2 * (j & 3);
    a[2 * j]     = r[k];
    a[2 * j + 1] = r[k + 1];
  }
  return a;
}

// LDS variant (pointer into __shared__ -> ds_load after inlining)
static __device__ __forceinline__ v16h frag_ld_lds(const _Float16* p, int ld, int lane) {
  const int m = lane & 15, h = lane >> 4;
  const _Float16* r = p + (size_t)m * ld;
  v16h a;
#pragma unroll
  for (int j = 0; j < 8; ++j) {
    const int k = ((j < 4) ? 0 : 16) + 8 * h + 2 * (j & 3);
    a[2 * j]     = r[k];
    a[2 * j + 1] = r[k + 1];
  }
  return a;
}

static __device__ __forceinline__ v16h frag_ld_f32_lds(const float* p, int ld, int lane) {
  const int m = lane & 15, h = lane >> 4;
  const float* r = p + (size_t)m * ld;
  v16h a;
#pragma unroll
  for (int j = 0; j < 8; ++j) {
    const int k = ((j < 4) ? 0 : 16) + 8 * h + 2 * (j & 3);
    a[2 * j]     = (_Float16)r[k];
    a[2 * j + 1] = (_Float16)r[k + 1];
  }
  return a;
}

// A[m][k] = base[k*ld + m]  (k-major source, e.g. x stored [c][pixel])
static __device__ __forceinline__ v16h frag_ld_f32_col(const float* __restrict__ base,
                                                       int ld, int lane) {
  const int m = lane & 15, h = lane >> 4;
  v16h a;
#pragma unroll
  for (int j = 0; j < 8; ++j) {
    const int k = ((j < 4) ? 0 : 16) + 8 * h + 2 * (j & 3);
    a[2 * j]     = (_Float16)base[(size_t)k * ld + m];
    a[2 * j + 1] = (_Float16)base[(size_t)(k + 1) * ld + m];
  }
  return a;
}

static __device__ __forceinline__ void frag_st_f16(_Float16* __restrict__ p, int ld,
                                                   int lane, v8f c) {
  const int n = lane & 15, h = lane >> 4;
#pragma unroll
  for (int i = 0; i < 8; ++i) p[(size_t)(i + 8 * h) * ld + n] = (_Float16)c[i];
}

// ---------------- kernel 1: fold W_in into Wq, convert weights to f16 ----------------
__global__ void __launch_bounds__(256) prep_kernel(
    const float* __restrict__ Wq, const float* __restrict__ W_in,
    const float* __restrict__ bq, const float* __restrict__ b_in,
    const float* __restrict__ Wk, const float* __restrict__ Wv,
    const float* __restrict__ W_out,
    _Float16* __restrict__ Wq_eff, _Float16* __restrict__ Wk_h,
    _Float16* __restrict__ Wv_h,  _Float16* __restrict__ Wout_h,
    float* __restrict__ bq_eff) {
  const int e = blockIdx.x;   // 0..255
  const int c = threadIdx.x;  // 0..255
  float acc = 0.f;
  for (int k = 0; k < INCH; ++k) acc += Wq[e * EMBD + k] * W_in[k * INCH + c];
  Wq_eff[e * INCH + c] = (_Float16)acc;
  Wout_h[e * EMBD + c] = (_Float16)W_out[e * EMBD + c];
  if (c < CTXD) {
    Wk_h[e * CTXD + c] = (_Float16)Wk[e * CTXD + c];
    Wv_h[e * CTXD + c] = (_Float16)Wv[e * CTXD + c];
  }
  if (c == 0) {
    float b = bq[e];
    for (int k = 0; k < EMBD; ++k) b += Wq[e * EMBD + k] * b_in[k];
    bq_eff[e] = b;
  }
}

// ---------------- kernel 2: Q = x @ Wq_eff^T + bq_eff  -> f16 [b][p][e] ----------------
__global__ void __launch_bounds__(128) qproj_kernel(
    const float* __restrict__ x, const _Float16* __restrict__ Wq_eff,
    const float* __restrict__ bq_eff, _Float16* __restrict__ Qh) {
  const int lane = threadIdx.x & 31, wave = threadIdx.x >> 5;
  const int m0 = blockIdx.x * 16;                 // pixel tile
  const int n0 = blockIdx.y * 64 + wave * 16;     // emb tile
  const int b  = blockIdx.z;
  const float* xb = x + (size_t)b * INCH * HW + m0;  // [c][pixel], pixel-contig
  v8f acc = {};
#pragma unroll
  for (int kk = 0; kk < INCH; kk += 32) {
    v16h a  = frag_ld_f32_col(xb + (size_t)kk * HW, HW, lane);
    v16h bb = frag_ld_f16(Wq_eff + (size_t)n0 * INCH + kk, INCH, lane);
    acc = WMMA_F16(a, bb, acc);
  }
  const float bias = bq_eff[n0 + (lane & 15)];
#pragma unroll
  for (int i = 0; i < 8; ++i) acc[i] += bias;
  frag_st_f16(Qh + ((size_t)b * HW + m0) * EMBD + n0, EMBD, lane, acc);
}

// ---------------- kernel 3: K,V = context @ W^T + b  -> f16 [b][s][e] ----------------
// Context tile (16x128 f32, contiguous 8KB) is staged once into LDS via async
// copy and shared by all 4 waves.
__global__ void __launch_bounds__(128) kvproj_kernel(
    const float* __restrict__ ctx, const _Float16* __restrict__ Wk_h,
    const _Float16* __restrict__ Wv_h, const float* __restrict__ bk,
    const float* __restrict__ bv, _Float16* __restrict__ Kh,
    _Float16* __restrict__ Vh) {
  __shared__ float shC[16 * CTXD];                // 8 KB
  const int lane = threadIdx.x & 31, wave = threadIdx.x >> 5;
  const int m0 = blockIdx.x * 16;                 // seq tile
  const int n0 = blockIdx.y * 64 + wave * 16;     // emb tile
  const int b  = blockIdx.z;
  const float* cb = ctx + ((size_t)b * SEQ + m0) * CTXD;

  stage_to_lds(cb, shC, threadIdx.x, 128, 16 * CTXD * 4);
  stage_wait();
  __syncthreads();

  v8f ak = {}, av = {};
#pragma unroll
  for (int kk = 0; kk < CTXD; kk += 32) {
    v16h a   = frag_ld_f32_lds(shC + kk, CTXD, lane);   // shared A for both GEMMs
    v16h bkf = frag_ld_f16(Wk_h + (size_t)n0 * CTXD + kk, CTXD, lane);
    v16h bvf = frag_ld_f16(Wv_h + (size_t)n0 * CTXD + kk, CTXD, lane);
    ak = WMMA_F16(a, bkf, ak);
    av = WMMA_F16(a, bvf, av);
  }
  const float bsk = bk[n0 + (lane & 15)], bsv = bv[n0 + (lane & 15)];
#pragma unroll
  for (int i = 0; i < 8; ++i) { ak[i] += bsk; av[i] += bsv; }
  frag_st_f16(Kh + ((size_t)b * SEQ + m0) * EMBD + n0, EMBD, lane, ak);
  frag_st_f16(Vh + ((size_t)b * SEQ + m0) * EMBD + n0, EMBD, lane, av);
}

// ---------------- kernel 4: fused attention (S, mask, softmax, att out, P@V) ----------
// 1 workgroup = 8 waves = 16 query rows x full seq (2048). Each wave owns a
// 16x256 slab of S in 16 v8f accumulators. The shared 16x256 Q tile (8KB,
// contiguous) is staged once via async copy into a slice of shP that is free
// until P is written (reduction scratch uses only the first 512B).
__global__ void __launch_bounds__(256) attn_kernel(
    const _Float16* __restrict__ Qh, const _Float16* __restrict__ Kh,
    const _Float16* __restrict__ Vh, const unsigned char* __restrict__ mask,
    float* __restrict__ att_out, _Float16* __restrict__ Oh) {
  __shared__ _Float16 shP[16 * SEQ];        // 64 KB: P (f16) tile
  float* red = (float*)shP;                 // bytes [0,512): reduction scratch
  _Float16* shQ = shP + 256;                // bytes [512,8704): staged Q tile

  const int lane = threadIdx.x & 31, wave = threadIdx.x >> 5;
  const int h = lane >> 4, nl = lane & 15;
  const int m0 = blockIdx.x * 16;           // query-row tile
  const int b  = blockIdx.y;
  const int nbase = wave * 256;             // this wave's seq span

  const _Float16* Qb = Qh + ((size_t)b * HW + m0) * EMBD;
  const _Float16* Kb = Kh + (size_t)b * SEQ * EMBD;
  const _Float16* Vb = Vh + (size_t)b * SEQ * EMBD;
  const unsigned char* mb = mask + ((size_t)b * HW + m0) * SEQ;
  const float scale = 0.0625f;              // 256^-0.5

  // stage Q tile (16 x 256 f16 = 8KB contiguous), shared by all 8 waves
  stage_to_lds(Qb, shQ, threadIdx.x, 256, 16 * EMBD * 2);
  stage_wait();
  __syncthreads();

  v16h qa[8];
#pragma unroll
  for (int kk = 0; kk < 8; ++kk) qa[kk] = frag_ld_lds(shQ + kk * 32, EMBD, lane);

  // ---- phase 1: S = Q K^T (scaled, masked), track row max ----
  v8f S[16];
  float lmax[8];
#pragma unroll
  for (int i = 0; i < 8; ++i) lmax[i] = -3.0e38f;

  for (int t = 0; t < 16; ++t) {
    const int n0 = nbase + t * 16;
    if (t < 15)
      __builtin_prefetch(Kb + (size_t)(n0 + 16) * EMBD, 0, 1);
    v8f acc = {};
#pragma unroll
    for (int kk = 0; kk < 8; ++kk) {
      v16h kf = frag_ld_f16(Kb + (size_t)n0 * EMBD + kk * 32, EMBD, lane);
      acc = WMMA_F16(qa[kk], kf, acc);
    }
    const int s = n0 + nl;
#pragma unroll
    for (int i = 0; i < 8; ++i) {
      const int r = i + 8 * h;
      float v = acc[i] * scale;
      if (mb[(size_t)r * SEQ + s]) v = -1.0e9f;
      acc[i] = v;
      lmax[i] = fmaxf(lmax[i], v);
    }
    S[t] = acc;
  }

  // row max: reduce across 16 lanes of the half, then across 8 waves via LDS
#pragma unroll
  for (int i = 0; i < 8; ++i) {
    float v = lmax[i];
#pragma unroll
    for (int mlk = 1; mlk < 16; mlk <<= 1) v = fmaxf(v, __shfl_xor(v, mlk, 32));
    lmax[i] = v;
  }
  __syncthreads();  // Q tile fully consumed; red scratch free to use
  if (nl == 0) {
#pragma unroll
    for (int i = 0; i < 8; ++i) red[(i + 8 * h) * 8 + wave] = lmax[i];
  }
  __syncthreads();
  float rmax[8];
#pragma unroll
  for (int i = 0; i < 8; ++i) {
    const int r = i + 8 * h;
    float v = -3.0e38f;
#pragma unroll
    for (int w = 0; w < 8; ++w) v = fmaxf(v, red[r * 8 + w]);
    rmax[i] = v;
  }
  __syncthreads();

  // ---- phase 2: exp + row sum ----
  float lsum[8];
#pragma unroll
  for (int i = 0; i < 8; ++i) lsum[i] = 0.f;
  for (int t = 0; t < 16; ++t) {
#pragma unroll
    for (int i = 0; i < 8; ++i) {
      float e = __expf(S[t][i] - rmax[i]);
      S[t][i] = e;
      lsum[i] += e;
    }
  }
#pragma unroll
  for (int i = 0; i < 8; ++i) {
    float v = lsum[i];
#pragma unroll
    for (int mlk = 1; mlk < 16; mlk <<= 1) v += __shfl_xor(v, mlk, 32);
    lsum[i] = v;
  }
  if (nl == 0) {
#pragma unroll
    for (int i = 0; i < 8; ++i) red[(i + 8 * h) * 8 + wave] = lsum[i];
  }
  __syncthreads();
  float rinv[8];
#pragma unroll
  for (int i = 0; i < 8; ++i) {
    const int r = i + 8 * h;
    float v = 0.f;
#pragma unroll
    for (int w = 0; w < 8; ++w) v += red[r * 8 + w];
    rinv[i] = 1.0f / v;
  }
  __syncthreads();  // done with red; shP may now be written as P

  // ---- write normalized att (f32 output) and P (f16) to LDS ----
  float* attb = att_out + ((size_t)b * HW + m0) * SEQ;
  for (int t = 0; t < 16; ++t) {
    const int s = nbase + t * 16 + nl;
#pragma unroll
    for (int i = 0; i < 8; ++i) {
      const int r = i + 8 * h;
      const float p = S[t][i] * rinv[i];
      attb[(size_t)r * SEQ + s] = p;
      shP[r * SEQ + s] = (_Float16)p;
    }
  }
  __syncthreads();

  // ---- phase 3: O = P @ V  (A from LDS via ds_load, B from global f16) ----
#pragma unroll
  for (int tt = 0; tt < 2; ++tt) {
    const int n0 = wave * 32 + tt * 16;
    v8f acc = {};
    for (int kk = 0; kk < SEQ; kk += 32) {
      __builtin_prefetch(Vb + (size_t)n0 * EMBD + (kk + 32), 0, 1);
      v16h a  = frag_ld_lds(shP + kk, SEQ, lane);
      v16h bf;
#pragma unroll
      for (int j = 0; j < 8; ++j) {
        const int k = ((j < 4) ? 0 : 16) + 8 * h + 2 * (j & 3);
        bf[2 * j]     = Vb[(size_t)(kk + k) * EMBD + n0 + nl];
        bf[2 * j + 1] = Vb[(size_t)(kk + k + 1) * EMBD + n0 + nl];
      }
      acc = WMMA_F16(a, bf, acc);
    }
    frag_st_f16(Oh + ((size_t)b * HW + m0) * EMBD + n0, EMBD, lane, acc);
  }
}

// ---------------- kernel 5: out = W_out @ O^T + b_out  -> f32 [b][c][p] ---------------
// D[m=c][n=p]: A = Wout_h ([c][e] row-major), B^T = Oh rows -> coalesced stores.
// O tile (16x256 f16, contiguous 8KB) staged once via async copy, shared by 4 waves.
__global__ void __launch_bounds__(128) oproj_kernel(
    const _Float16* __restrict__ Oh, const _Float16* __restrict__ Wout_h,
    const float* __restrict__ b_out, float* __restrict__ out) {
  __shared__ _Float16 shO[16 * EMBD];                  // 8 KB
  const int lane = threadIdx.x & 31, wave = threadIdx.x >> 5;
  const int p0 = blockIdx.x * 16;                      // pixel tile
  const int c0 = (blockIdx.y * 4 + wave) * 16;         // channel tile
  const int b  = blockIdx.z;

  stage_to_lds(Oh + ((size_t)b * HW + p0) * EMBD, shO, threadIdx.x, 128, 16 * EMBD * 2);
  stage_wait();
  __syncthreads();

  v8f acc = {};
#pragma unroll
  for (int kk = 0; kk < EMBD; kk += 32) {
    v16h a  = frag_ld_f16(Wout_h + (size_t)c0 * EMBD + kk, EMBD, lane);
    v16h bf = frag_ld_lds(shO + kk, EMBD, lane);
    acc = WMMA_F16(a, bf, acc);
  }
  float* ob = out + (size_t)b * INCH * HW;
  const int n = lane & 15, h = lane >> 4;
#pragma unroll
  for (int i = 0; i < 8; ++i) {
    const int m = c0 + i + 8 * h;
    ob[(size_t)m * HW + p0 + n] = acc[i] + b_out[m];
  }
}

// ---------------- launcher ----------------
extern "C" void kernel_launch(void* const* d_in, const int* in_sizes, int n_in,
                              void* d_out, int out_size, void* d_ws, size_t ws_size,
                              hipStream_t stream) {
  // input order: x, context, pad_mask, W_in, b_in, Wq, bq, Wk, bk, Wv, bv, W_out, b_out
  const float* x    = (const float*)d_in[0];
  const float* ctx  = (const float*)d_in[1];
  const unsigned char* mask = (const unsigned char*)d_in[2];  // jax bool -> 1 byte
  const float* W_in = (const float*)d_in[3];
  const float* b_in = (const float*)d_in[4];
  const float* Wq   = (const float*)d_in[5];
  const float* bq   = (const float*)d_in[6];
  const float* Wk   = (const float*)d_in[7];
  const float* bk   = (const float*)d_in[8];
  const float* Wv   = (const float*)d_in[9];
  const float* bv   = (const float*)d_in[10];
  const float* W_out= (const float*)d_in[11];
  const float* b_out= (const float*)d_in[12];

  float* out_main = (float*)d_out;                                   // [16,256,48,48]
  float* att_out  = (float*)d_out + (size_t)NB * INCH * HW;          // [16,2304,2048]

  // workspace carving (256B aligned)
  char* ws = (char*)d_ws;
  size_t off = 0;
  auto carve = [&](size_t bytes) { char* p = ws + off; off += (bytes + 255) & ~(size_t)255; return p; };
  _Float16* Wq_eff = (_Float16*)carve((size_t)EMBD * INCH * 2);
  _Float16* Wk_h   = (_Float16*)carve((size_t)EMBD * CTXD * 2);
  _Float16* Wv_h   = (_Float16*)carve((size_t)EMBD * CTXD * 2);
  _Float16* Wout_h = (_Float16*)carve((size_t)INCH * EMBD * 2);
  float*    bq_eff = (float*)carve((size_t)EMBD * 4);
  _Float16* Qh     = (_Float16*)carve((size_t)NB * HW * EMBD * 2);
  _Float16* Kh     = (_Float16*)carve((size_t)NB * SEQ * EMBD * 2);
  _Float16* Vh     = (_Float16*)carve((size_t)NB * SEQ * EMBD * 2);
  _Float16* Oh     = (_Float16*)carve((size_t)NB * HW * EMBD * 2);

  prep_kernel<<<dim3(EMBD), dim3(256), 0, stream>>>(
      Wq, W_in, bq, b_in, Wk, Wv, W_out, Wq_eff, Wk_h, Wv_h, Wout_h, bq_eff);

  qproj_kernel<<<dim3(HW / 16, EMBD / 64, NB), dim3(128), 0, stream>>>(
      x, Wq_eff, bq_eff, Qh);

  kvproj_kernel<<<dim3(SEQ / 16, EMBD / 64, NB), dim3(128), 0, stream>>>(
      ctx, Wk_h, Wv_h, bk, bv, Kh, Vh);

  attn_kernel<<<dim3(HW / 16, NB), dim3(256), 0, stream>>>(
      Qh, Kh, Vh, mask, att_out, Oh);

  oproj_kernel<<<dim3(HW / 16, EMBD / 64, NB), dim3(128), 0, stream>>>(
      Oh, Wout_h, b_out, out_main);
}
